// CPIA_SA_80771154968957
// MI455X (gfx1250) — matrix operand order repaired and checked by reading, not compiled, not run on hardware
//
#include <hip/hip_runtime.h>
#include <hip/hip_bf16.h>
#include <math.h>

// ---------------------------------------------------------------------------
// CPIA-SA block for MI455X (gfx1250). 1x1 convs -> WMMA f16 GEMMs with
// TDM (tensor_load_to_lds) A-tile staging and b128-vectorized LDS fragments.
// ---------------------------------------------------------------------------

typedef __attribute__((ext_vector_type(16))) _Float16 v16h;
typedef __attribute__((ext_vector_type(8)))  _Float16 v8h;
typedef __attribute__((ext_vector_type(8)))  float    v8f;
typedef __attribute__((ext_vector_type(4)))  unsigned uv4;
typedef __attribute__((ext_vector_type(8)))  int      iv8;
typedef __attribute__((ext_vector_type(4)))  int      iv4;

#define BB    8
#define HEADS 8
#define CH    32          // per-head channel dim
#define HWN   4096        // 64*64
#define WW    64
#define HH    64

#if defined(__has_builtin)
#if __has_builtin(__builtin_amdgcn_tensor_load_to_lds) && __has_builtin(__builtin_amdgcn_s_wait_tensorcnt)
#define USE_TDM 1
#endif
#endif

// f32 -> f16 weight pre-conversion
__global__ __launch_bounds__(256) void f32_to_f16(const float* __restrict__ src,
                                                  _Float16* __restrict__ dst, long n)
{
  const long i = (long)blockIdx.x * 256 + threadIdx.x;
  if (i < n) dst[i] = (_Float16)src[i];
}

// ===========================================================================
// WMMA GEMM:  Y[b] = act( A16[M,K] @ X[b][K,4096] + bias )
// block = 256 threads (8 waves), block tile 64x64, K step 32.
// A tile (f16, 64x32) staged via TDM tensor_load_to_lds (wave 0 issues).
// B tile converted f32->f16 and stored transposed [n][k] for b128 reads.
// Requires M%64==0, K%32==0. N fixed = 4096.
// ===========================================================================
__global__ __launch_bounds__(256) void gemm_wmma_f16(
    const _Float16* __restrict__ A16, const float* __restrict__ X,
    float* __restrict__ Y, const float* __restrict__ bias,
    int M, int K, long strideX, long strideY, int act)
{
  __shared__ _Float16 As[64 * 32];    // [m][k] row-major (TDM fill order)
  __shared__ _Float16 BsT[64 * 40];   // [n][k], padded to 40 halfs/row

  const int tid   = threadIdx.x;
  const int lane  = tid & 31;
  const int wave  = tid >> 5;        // 0..7
  const int wm    = wave & 3;        // 16-row sub-tile
  const int wn    = wave >> 2;       // 32-col half
  const int l15   = lane & 15;
  const int lhalf = lane >> 4;

  const int n0 = blockIdx.x * 64;
  const int m0 = blockIdx.y * 64;
  const int b  = blockIdx.z;

  const int bc = tid >> 2;           // B tile: column 0..63
  const int bt = tid & 3;            // B tile: k-chunk 0..3

  v8f acc0 = {0.f,0.f,0.f,0.f,0.f,0.f,0.f,0.f};
  v8f acc1 = {0.f,0.f,0.f,0.f,0.f,0.f,0.f,0.f};

#ifdef USE_TDM
  const unsigned ldsA = (unsigned)(size_t)(void*)As;
#endif

  for (int k0 = 0; k0 < K; k0 += 32) {
    __syncthreads();

    // ---- A tile: 64x32 f16 ----
#ifdef USE_TDM
    if (wave == 0) {
      const unsigned long long ga =
          (unsigned long long)(const void*)(A16 + (size_t)m0 * K + k0);
      uv4 g0;
      g0[0] = 1u;                                       // count=1 (valid D#)
      g0[1] = ldsA;                                     // lds_addr
      g0[2] = (unsigned)(ga & 0xFFFFFFFFu);             // global_addr[31:0]
      g0[3] = (unsigned)((ga >> 32) & 0x01FFFFFFu) | (2u << 30); // addr[56:32], type=2
      iv8 g1;
      g1[0] = (int)(1u << 16);                          // data_size=1 (2 bytes)
      g1[1] = (int)(((unsigned)K & 0xFFFFu) << 16);     // tensor_dim0 lo16
      g1[2] = (int)((((unsigned)K >> 16) & 0xFFFFu) |
                    (((unsigned)M & 0xFFFFu) << 16));   // dim0 hi | dim1 lo
      g1[3] = (int)((((unsigned)M >> 16) & 0xFFFFu) |
                    (32u << 16));                       // dim1 hi | tile_dim0=32
      g1[4] = 64;                                       // tile_dim1=64, tile_dim2=0
      g1[5] = K;                                        // tensor_dim0_stride lo32
      g1[6] = 0;
      g1[7] = 0;
      iv4 gz4 = {0, 0, 0, 0};
      iv8 gz8 = {0, 0, 0, 0, 0, 0, 0, 0};
      __builtin_amdgcn_tensor_load_to_lds(g0, g1, gz4, gz4, gz8, 0);
    }
#else
    {
      const int r = tid >> 2, t = tid & 3;
      *(v8h*)(As + r * 32 + t * 8) =
          *(const v8h*)(A16 + (size_t)(m0 + r) * K + k0 + t * 8);
    }
#endif

    // ---- B tile: 32x64 f32 -> f16, stored transposed [n][k] ----
    {
      const float* xp = X + (size_t)b * strideX +
                        (size_t)(k0 + bt * 8) * HWN + n0 + bc;
      v8h tmp;
      #pragma unroll
      for (int j = 0; j < 8; ++j) tmp[j] = (_Float16)xp[(size_t)j * HWN];
      *(v8h*)(BsT + bc * 40 + bt * 8) = tmp;
    }

#ifdef USE_TDM
    if (wave == 0) __builtin_amdgcn_s_wait_tensorcnt(0);
#endif
    __syncthreads();

    // ---- fragments (b128 LDS reads) ----
    const int arow = 16 * wm + l15;
    const v8h* ap = (const v8h*)(As + arow * 32);
    const v8h alo = ap[lhalf];          // K 0..7 | 8..15
    const v8h ahi = ap[2 + lhalf];      // K 16..23 | 24..31
    v16h afrag;
    #pragma unroll
    for (int j = 0; j < 8; ++j) { afrag[j] = alo[j]; afrag[8 + j] = ahi[j]; }

    const int col0 = 32 * wn + l15;
    const v8h* bp0 = (const v8h*)(BsT + col0 * 40);
    const v8h* bp1 = (const v8h*)(BsT + (col0 + 16) * 40);
    const v8h b0a = bp0[lhalf], b0b = bp0[2 + lhalf];
    const v8h b1a = bp1[lhalf], b1b = bp1[2 + lhalf];
    v16h bf0, bf1;
    #pragma unroll
    for (int j = 0; j < 8; ++j) {
      bf0[j] = b0a[j]; bf0[8 + j] = b0b[j];
      bf1[j] = b1a[j]; bf1[8 + j] = b1b[j];
    }

    acc0 = __builtin_amdgcn_wmma_f32_16x16x32_f16(false, afrag, false, bf0,
                                                  (short)0, acc0, false, false);
    acc1 = __builtin_amdgcn_wmma_f32_16x16x32_f16(false, afrag, false, bf1,
                                                  (short)0, acc1, false, false);
  }

  // epilogue: D layout -> global
  const int rbase = 16 * wm + (lhalf ? 8 : 0);
  #pragma unroll
  for (int r = 0; r < 8; ++r) {
    const int row = m0 + rbase + r;
    const float bv = bias ? bias[row] : 0.f;
    const int colA = n0 + 32 * wn + l15;
    float o0 = acc0[r] + bv;
    float o1 = acc1[r] + bv;
    if (act == 1) { o0 = fmaxf(o0, 0.f); o1 = fmaxf(o1, 0.f); }
    Y[(size_t)b * strideY + (size_t)row * HWN + colA]      = o0;
    Y[(size_t)b * strideY + (size_t)row * HWN + colA + 16] = o1;
  }
}

// ===========================================================================
// depthwise 3x3, zero pad, out channel oc reads in channel oc/growth
// ===========================================================================
__global__ __launch_bounds__(256) void dwconv3x3(
    const float* __restrict__ in, long inBatchStride, int Cout, int growth,
    const float* __restrict__ w, const float* __restrict__ bias,
    float* __restrict__ out)
{
  const long idx = (long)blockIdx.x * 256 + threadIdx.x;
  const int  s   = (int)(idx & (HWN - 1));
  const int  oc  = (int)((idx >> 12) % Cout);
  const int  b   = (int)(idx / ((long)Cout * HWN));
  const int  y0  = s >> 6, x0 = s & 63;
  const float* ib = in + (size_t)b * inBatchStride + (size_t)(oc / growth) * HWN;
  float acc = bias ? bias[oc] : 0.f;
  #pragma unroll
  for (int ky = 0; ky < 3; ++ky) {
    const int yy = y0 + ky - 1;
    if (yy < 0 || yy >= HH) continue;
    #pragma unroll
    for (int kx = 0; kx < 3; ++kx) {
      const int xx = x0 + kx - 1;
      if (xx < 0 || xx >= WW) continue;
      acc += w[oc * 9 + ky * 3 + kx] * ib[yy * WW + xx];
    }
  }
  out[idx] = acc;
}

// per-(b,ch) spatial mean
__global__ __launch_bounds__(256) void chan_mean(const float* __restrict__ in,
                                                 float* __restrict__ out)
{
  const size_t base = (size_t)blockIdx.x * HWN;
  float s = 0.f;
  for (int i = threadIdx.x; i < HWN; i += 256) s += in[base + i];
  __shared__ float red[256];
  red[threadIdx.x] = s; __syncthreads();
  for (int o = 128; o > 0; o >>= 1) {
    if (threadIdx.x < o) red[threadIdx.x] += red[threadIdx.x + o];
    __syncthreads();
  }
  if (threadIdx.x == 0) out[blockIdx.x] = red[0] * (1.f / HWN);
}

// softmax over 512 channels per batch
__global__ __launch_bounds__(512) void softmax512(const float* __restrict__ in,
                                                  float* __restrict__ out)
{
  const int b = blockIdx.x;
  __shared__ float red[512];
  const float v = in[b * 512 + threadIdx.x];
  red[threadIdx.x] = v; __syncthreads();
  for (int o = 256; o > 0; o >>= 1) {
    if (threadIdx.x < o) red[threadIdx.x] = fmaxf(red[threadIdx.x], red[threadIdx.x + o]);
    __syncthreads();
  }
  const float mx = red[0]; __syncthreads();
  const float e = __expf(v - mx);
  red[threadIdx.x] = e; __syncthreads();
  for (int o = 256; o > 0; o >>= 1) {
    if (threadIdx.x < o) red[threadIdx.x] += red[threadIdx.x + o];
    __syncthreads();
  }
  out[b * 512 + threadIdx.x] = e / red[0];
}

// t = gelu_exact(ssoft[b,ch] * s[b,ch,hw])
__global__ __launch_bounds__(256) void gelu_gate512(const float* __restrict__ s,
                                                    const float* __restrict__ ssoft,
                                                    float* __restrict__ out)
{
  const long idx = (long)blockIdx.x * 256 + threadIdx.x;
  const int ch = (int)((idx >> 12) & 511);
  const int b  = (int)(idx >> 21);
  const float v = ssoft[b * 512 + ch] * s[idx];
  out[idx] = 0.5f * v * (1.f + erff(v * 0.70710678118f));
}

// L2-normalize q and k rows (channels 0..511 of the 768-ch qkv buffer), in place
__global__ __launch_bounds__(256) void l2_normalize_qk(float* __restrict__ qk)
{
  const int blk = blockIdx.x;                  // b*512 + cc
  const int b = blk >> 9, cc = blk & 511;
  float* p = qk + ((size_t)b * 768 + cc) * HWN;
  float s = 0.f;
  for (int i = threadIdx.x; i < HWN; i += 256) { const float v = p[i]; s += v * v; }
  __shared__ float red[256];
  red[threadIdx.x] = s; __syncthreads();
  for (int o = 128; o > 0; o >>= 1) {
    if (threadIdx.x < o) red[threadIdx.x] += red[threadIdx.x + o];
    __syncthreads();
  }
  const float inv = 1.f / fmaxf(sqrtf(red[0]), 1e-12f);
  for (int i = threadIdx.x; i < HWN; i += 256) p[i] *= inv;
}

// gate second 1x1 + sigmoid, block-reduced sum -> atomic into gsum
__global__ __launch_bounds__(256) void gate_reduce(const float* __restrict__ gh,
                                                   const float* __restrict__ w2,
                                                   const float* __restrict__ b2,
                                                   float* __restrict__ gsum)
{
  const int blk = blockIdx.x;                  // b*16 + chunk
  const int b = blk >> 4;
  const int s = ((blk & 15) << 8) + threadIdx.x;
  const float* gb = gh + (size_t)b * 128 * HWN;
  float acc = b2[0];
  for (int c = 0; c < 128; ++c) acc += w2[c] * gb[(size_t)c * HWN + s];
  const float g = 1.f / (1.f + __expf(-acc));
  __shared__ float red[256];
  red[threadIdx.x] = g; __syncthreads();
  for (int o = 128; o > 0; o >>= 1) {
    if (threadIdx.x < o) red[threadIdx.x] += red[threadIdx.x + o];
    __syncthreads();
  }
  if (threadIdx.x == 0) atomicAdd(gsum, red[0]);
}

__global__ void zero_scalars(float* gsum) { *gsum = 0.f; }

__global__ void compute_dynk(const float* __restrict__ gsum, int* __restrict__ dynk)
{
  *dynk = (int)(32.f * (*gsum) / 32768.f);     // C * mean(g), truncation
}

// attn_raw[bh][c][d] = temperature[h] * sum_n q[c][n] k[d][n]
__global__ __launch_bounds__(256) void attn_gram(const float* __restrict__ qk,
                                                 const float* __restrict__ temp,
                                                 float* __restrict__ raw)
{
  const int bh = blockIdx.x;
  const int b = bh >> 3, h = bh & 7;
  const float* qb = qk + ((size_t)b * 768 + h * CH) * HWN;
  const float* kb = qb + (size_t)256 * HWN;
  __shared__ float qs[32 * 64], ks[32 * 64];
  const int tid = threadIdx.x;
  float acc[4] = {0.f, 0.f, 0.f, 0.f};
  for (int n0 = 0; n0 < HWN; n0 += 64) {
    __syncthreads();
    for (int i = tid; i < 2048; i += 256) {
      const int r = i >> 6, c = i & 63;
      qs[i] = qb[(size_t)r * HWN + n0 + c];
      ks[i] = kb[(size_t)r * HWN + n0 + c];
    }
    __syncthreads();
    #pragma unroll
    for (int i = 0; i < 4; ++i) {
      const int e = tid + 256 * i;
      const int cc = e >> 5, dd = e & 31;
      float s = acc[i];
      for (int nn = 0; nn < 64; ++nn) s += qs[cc * 64 + nn] * ks[dd * 64 + nn];
      acc[i] = s;
    }
  }
  const float t = temp[h];
  #pragma unroll
  for (int i = 0; i < 4; ++i)
    raw[(size_t)bh * 1024 + tid + 256 * i] = acc[i] * t;
}

// rank-based top-k mask + row softmax (stable-argsort rank semantics)
__global__ __launch_bounds__(1024) void attn_topk_softmax(const float* __restrict__ raw,
                                                          const int* __restrict__ dynk_p,
                                                          float* __restrict__ fin)
{
  const int bh = blockIdx.x, tid = threadIdx.x;
  const int c = tid >> 5, d = tid & 31;
  __shared__ float sa[1024], sm_[1024];
  sa[tid] = raw[(size_t)bh * 1024 + tid];
  __syncthreads();
  const int dynk = *dynk_p;
  const float v = sa[c * 32 + d];
  int rank = 0;
  for (int j = 0; j < 32; ++j) {
    const float aj = sa[c * 32 + j];
    rank += (aj > v) || (aj == v && j < d);
  }
  sm_[tid] = (rank < dynk) ? v : -INFINITY;
  __syncthreads();
  float mx = -INFINITY;
  for (int j = 0; j < 32; ++j) mx = fmaxf(mx, sm_[c * 32 + j]);
  float den = 0.f;
  for (int j = 0; j < 32; ++j) den += __expf(sm_[c * 32 + j] - mx);
  fin[(size_t)bh * 1024 + tid] = __expf(sm_[tid] - mx) / den;
}

// out_att[b, h*32+c, :] = scale * attn_row @ V
__global__ __launch_bounds__(256) void attn_apply(const float* __restrict__ fin,
                                                  const float* __restrict__ qk,
                                                  const float* __restrict__ a1,
                                                  const float* __restrict__ a2,
                                                  const float* __restrict__ a3,
                                                  const float* __restrict__ a4,
                                                  float* __restrict__ out_att)
{
  const int blk = blockIdx.x;                  // b*256 + h*32 + c
  const int b = blk >> 8, hc = blk & 255;
  const int h = hc >> 5, c = hc & 31;
  __shared__ float arow[32];
  if (threadIdx.x < 32)
    arow[threadIdx.x] = fin[((size_t)(b * 8 + h) * 32 + c) * 32 + threadIdx.x];
  __syncthreads();
  const float scale = a1[0] + a2[0] + a3[0] + a4[0];
  const float* vb = qk + ((size_t)b * 768 + 512 + h * CH) * HWN;
  float* ob = out_att + ((size_t)b * 256 + hc) * HWN;
  for (int n = threadIdx.x; n < HWN; n += 256) {
    float s = 0.f;
    #pragma unroll
    for (int d = 0; d < 32; ++d) s += arow[d] * vb[(size_t)d * HWN + n];
    ob[n] = s * scale;
  }
}

// channel interaction: cmean[b,256] -> 64 (BN,gelu) -> cmv[b,256]
__global__ __launch_bounds__(256) void ci_kernel(const float* __restrict__ cmean,
    const float* __restrict__ w1, const float* __restrict__ b1,
    const float* __restrict__ bng, const float* __restrict__ bnb,
    const float* __restrict__ bnm, const float* __restrict__ bnv,
    const float* __restrict__ w2, const float* __restrict__ b2,
    float* __restrict__ cmv)
{
  const int b = blockIdx.x, tid = threadIdx.x;
  __shared__ float cml[256], hid[64];
  cml[tid] = cmean[b * 256 + tid];
  __syncthreads();
  if (tid < 64) {
    float acc = b1[tid];
    for (int c = 0; c < 256; ++c) acc += w1[tid * 256 + c] * cml[c];
    const float z = (acc - bnm[tid]) * rsqrtf(bnv[tid] + 1e-5f) * bng[tid] + bnb[tid];
    hid[tid] = 0.5f * z * (1.f + erff(z * 0.70710678118f));
  }
  __syncthreads();
  float acc = b2[tid];
  for (int j = 0; j < 64; ++j) acc += w2[tid * 64 + j] * hid[j];
  cmv[b * 256 + tid] = acc;
}

// spatial interaction: y_d -> 32 (BN,gelu) -> 1 channel map smap[b,hw]
__global__ __launch_bounds__(256) void si_kernel(const float* __restrict__ yd,
    const float* __restrict__ w1, const float* __restrict__ b1,
    const float* __restrict__ bng, const float* __restrict__ bnb,
    const float* __restrict__ bnm, const float* __restrict__ bnv,
    const float* __restrict__ w2, const float* __restrict__ b2,
    float* __restrict__ smap)
{
  const int blk = blockIdx.x;                  // b*16 + chunk
  const int b = blk >> 4;
  const int s = ((blk & 15) << 8) + threadIdx.x;
  __shared__ float wl[32 * 256];
  for (int i = threadIdx.x; i < 8192; i += 256) wl[i] = w1[i];
  __syncthreads();
  float acc[32];
  #pragma unroll
  for (int j = 0; j < 32; ++j) acc[j] = b1[j];
  const float* yb = yd + (size_t)b * 256 * HWN + s;
  for (int c = 0; c < 256; ++c) {
    const float xv = yb[(size_t)c * HWN];
    #pragma unroll
    for (int j = 0; j < 32; ++j) acc[j] += wl[j * 256 + c] * xv;
  }
  float out = b2[0];
  #pragma unroll
  for (int j = 0; j < 32; ++j) {
    const float z = (acc[j] - bnm[j]) * rsqrtf(bnv[j] + 1e-5f) * bng[j] + bnb[j];
    out += w2[j] * (0.5f * z * (1.f + erff(z * 0.70710678118f)));
  }
  smap[b * HWN + s] = out;
}

// build proj input: [att*sig(smap) ; y_d*sig(cmv)]
__global__ __launch_bounds__(256) void combine_kernel(const float* __restrict__ out_att,
                                                      const float* __restrict__ yd,
                                                      const float* __restrict__ smap,
                                                      const float* __restrict__ cmv,
                                                      float* __restrict__ proj_in)
{
  const long idx = (long)blockIdx.x * 256 + threadIdx.x;   // < 8*256*4096
  const int s  = (int)(idx & (HWN - 1));
  const int ch = (int)((idx >> 12) & 255);
  const int b  = (int)(idx >> 20);
  const float sg = 1.f / (1.f + __expf(-smap[b * HWN + s]));
  const float cg = 1.f / (1.f + __expf(-cmv[b * 256 + ch]));
  proj_in[((size_t)b * 512 + ch) * HWN + s]       = out_att[idx] * sg;
  proj_in[((size_t)b * 512 + 256 + ch) * HWN + s] = yd[idx] * cg;
}

// 64-point DFT along rows (alongW=1) or columns (alongW=0); sign=-1 fwd, +1 inv
__global__ __launch_bounds__(64) void dft64(const float* __restrict__ inRe,
                                            const float* __restrict__ inIm,
                                            float* __restrict__ outRe,
                                            float* __restrict__ outIm,
                                            int alongW, float sign, float scale)
{
  const int img = blockIdx.x >> 6;             // b*512 + ch
  const int li  = blockIdx.x & 63;
  const size_t base  = (size_t)img * HWN;
  const size_t lbase = base + (alongW ? (size_t)li * WW : (size_t)li);
  const size_t stride = alongW ? 1 : WW;
  __shared__ float xr[64], xi[64];
  const int k = threadIdx.x;
  xr[k] = inRe[lbase + (size_t)k * stride];
  xi[k] = inIm ? inIm[lbase + (size_t)k * stride] : 0.f;
  __syncthreads();
  const float w0 = sign * 6.28318530717958647f * (float)k / 64.f;
  float sr = 0.f, sj = 0.f;
  for (int j = 0; j < 64; ++j) {
    float s, c;
    __sincosf(w0 * (float)j, &s, &c);
    sr += xr[j] * c - xi[j] * s;
    sj += xr[j] * s + xi[j] * c;
  }
  outRe[lbase + (size_t)k * stride] = sr * scale;
  if (outIm) outIm[lbase + (size_t)k * stride] = sj * scale;
}

// per-channel scale on [real;imag]
__global__ __launch_bounds__(256) void fft_scale(float* __restrict__ re,
                                                 float* __restrict__ im,
                                                 const float* __restrict__ w,
                                                 const float* __restrict__ bb)
{
  const long idx = (long)blockIdx.x * 256 + threadIdx.x;   // < 8*512*4096
  const int ch = (int)((idx >> 12) & 511);
  re[idx] = re[idx] * w[ch]       + bb[ch];
  im[idx] = im[idx] * w[512 + ch] + bb[512 + ch];
}

// ===========================================================================
extern "C" void kernel_launch(void* const* d_in, const int* in_sizes, int n_in,
                              void* d_out, int out_size, void* d_ws, size_t ws_size,
                              hipStream_t stream)
{
  (void)in_sizes; (void)n_in; (void)out_size; (void)ws_size;
  const float* x        = (const float*)d_in[0];
  const float* lin0_w   = (const float*)d_in[1];
  const float* lin0_b   = (const float*)d_in[2];
  const float* c0a_w    = (const float*)d_in[3];
  const float* c0a_b    = (const float*)d_in[4];
  const float* c0b_w    = (const float*)d_in[5];
  const float* c0b_b    = (const float*)d_in[6];
  const float* c1_w     = (const float*)d_in[7];
  const float* c1_b     = (const float*)d_in[8];
  const float* qkv_w    = (const float*)d_in[9];
  const float* qkvdw_w  = (const float*)d_in[10];
  const float* proj_w   = (const float*)d_in[11];
  const float* temp     = (const float*)d_in[12];
  const float* a1       = (const float*)d_in[13];
  const float* a2       = (const float*)d_in[14];
  const float* a3       = (const float*)d_in[15];
  const float* a4       = (const float*)d_in[16];
  const float* ci_c1_w  = (const float*)d_in[17];
  const float* ci_c1_b  = (const float*)d_in[18];
  const float* ci_bn_g  = (const float*)d_in[19];
  const float* ci_bn_b  = (const float*)d_in[20];
  const float* ci_bn_m  = (const float*)d_in[21];
  const float* ci_bn_v  = (const float*)d_in[22];
  const float* ci_c2_w  = (const float*)d_in[23];
  const float* ci_c2_b  = (const float*)d_in[24];
  const float* si_c1_w  = (const float*)d_in[25];
  const float* si_c1_b  = (const float*)d_in[26];
  const float* si_bn_g  = (const float*)d_in[27];
  const float* si_bn_b  = (const float*)d_in[28];
  const float* si_bn_m  = (const float*)d_in[29];
  const float* si_bn_v  = (const float*)d_in[30];
  const float* si_c2_w  = (const float*)d_in[31];
  const float* si_c2_b  = (const float*)d_in[32];
  const float* g_c1_w   = (const float*)d_in[33];
  const float* g_c1_b   = (const float*)d_in[34];
  const float* g_c2_w   = (const float*)d_in[35];
  const float* g_c2_b   = (const float*)d_in[36];
  const float* fft_w    = (const float*)d_in[37];
  const float* fft_b    = (const float*)d_in[38];
  float* out = (float*)d_out;

  // ---- workspace arena (floats); peak ~507 MB ----
  float* ws = (float*)d_ws;
  const size_t SZ512 = (size_t)BB * 512 * HWN;     // 16 Mfloats
  const size_t SZ256 = (size_t)BB * 256 * HWN;     //  8 Mfloats
  const size_t SZ768 = (size_t)BB * 768 * HWN;     // 24 Mfloats
  float* L     = ws;                 // lin0 out; later x_out
  float* S1    = L + SZ512;          // gate hid -> dwconv -> gelu-gate -> out_att -> F1re
  float* S2    = S1 + SZ512;         // c0b out -> F1im
  float* YD    = S2 + SZ512;         // y_d
  float* QKV   = YD + SZ256;         // qkv 1x1; later F2re
  float* QKV2  = QKV + SZ768;        // qkv dw (q,k,v); later F2im
  float* PROJ  = QKV2 + SZ768;       // proj input (concat)
  float* SMALL = PROJ + SZ512;
  float* smean = SMALL;              // 4096
  float* ssoft = smean + 4096;       // 4096
  float* gsum  = ssoft + 4096;       // 1
  int*   dynk  = (int*)(gsum + 2);   // 1 int
  float* araw  = gsum + 4;           // 65536
  float* afin  = araw + 65536;       // 65536
  float* cmean = afin + 65536;       // 2048
  float* cmv   = cmean + 2048;       // 2048
  float* smap  = cmv + 2048;         // 32768
  float* OATT  = S1;                 // out_att reuses S1
  float* F1re = S1, *F1im = S2, *F2re = QKV, *F2im = QKV2;

  // f16 weight arena (16-byte aligned)
  _Float16* WH = (_Float16*)(((size_t)(smap + 32768) + 15) & ~(size_t)15);
  _Float16* lin0h = WH;                 // 262144
  _Float16* c0bh  = lin0h + 262144;     // 262144
  _Float16* c1h   = c0bh + 262144;      // 131072
  _Float16* qkvh  = c1h + 131072;       // 196608
  _Float16* g1h   = qkvh + 196608;      // 32768
  _Float16* projh = g1h + 32768;        // 262144

  const long st512 = (long)512 * HWN, st256 = (long)256 * HWN;
  const long st768 = (long)768 * HWN, st128 = (long)128 * HWN;
  const dim3 blk256(256);

  hipLaunchKernelGGL(zero_scalars, dim3(1), dim3(1), 0, stream, gsum);
  // weight pre-conversion f32 -> f16
  hipLaunchKernelGGL(f32_to_f16, dim3(1024), blk256, 0, stream, lin0_w, lin0h, 262144L);
  hipLaunchKernelGGL(f32_to_f16, dim3(1024), blk256, 0, stream, c0b_w,  c0bh,  262144L);
  hipLaunchKernelGGL(f32_to_f16, dim3(512),  blk256, 0, stream, c1_w,   c1h,   131072L);
  hipLaunchKernelGGL(f32_to_f16, dim3(768),  blk256, 0, stream, qkv_w,  qkvh,  196608L);
  hipLaunchKernelGGL(f32_to_f16, dim3(128),  blk256, 0, stream, g_c1_w, g1h,   32768L);
  hipLaunchKernelGGL(f32_to_f16, dim3(1024), blk256, 0, stream, proj_w, projh, 262144L);

  // 1) lin0 -> L  (y = ch0..255, x_in = ch256..511)
  hipLaunchKernelGGL(gemm_wmma_f16, dim3(64, 8, 8), blk256, 0, stream,
                     lin0h, x, L, lin0_b, 512, 512, st512, st512, 0);
  // 2) gate: relu(conv1x1(x_in)) -> S1 ; reduce -> gsum ; dyn_k
  hipLaunchKernelGGL(gemm_wmma_f16, dim3(64, 2, 8), blk256, 0, stream,
                     g1h, L + (size_t)256 * HWN, S1, g_c1_b, 128, 256, st512, st128, 1);
  hipLaunchKernelGGL(gate_reduce, dim3(128), blk256, 0, stream, S1, g_c2_w, g_c2_b, gsum);
  hipLaunchKernelGGL(compute_dynk, dim3(1), dim3(1), 0, stream, gsum, dynk);
  // 3) spr branch
  hipLaunchKernelGGL(dwconv3x3, dim3(65536), blk256, 0, stream,
                     L, st512, 512, 2, c0a_w, c0a_b, S1);
  hipLaunchKernelGGL(gemm_wmma_f16, dim3(64, 8, 8), blk256, 0, stream,
                     c0bh, S1, S2, c0b_b, 512, 512, st512, st512, 0);
  hipLaunchKernelGGL(chan_mean, dim3(BB * 512), blk256, 0, stream, S2, smean);
  hipLaunchKernelGGL(softmax512, dim3(BB), dim3(512), 0, stream, smean, ssoft);
  hipLaunchKernelGGL(gelu_gate512, dim3(65536), blk256, 0, stream, S2, ssoft, S1);
  hipLaunchKernelGGL(gemm_wmma_f16, dim3(64, 4, 8), blk256, 0, stream,
                     c1h, S1, YD, c1_b, 256, 512, st512, st256, 0);
  // 4) qkv
  hipLaunchKernelGGL(gemm_wmma_f16, dim3(64, 12, 8), blk256, 0, stream,
                     qkvh, L + (size_t)256 * HWN, QKV, nullptr, 768, 256, st512, st768, 0);
  hipLaunchKernelGGL(dwconv3x3, dim3(98304), blk256, 0, stream,
                     QKV, st768, 768, 1, qkvdw_w, nullptr, QKV2);
  hipLaunchKernelGGL(l2_normalize_qk, dim3(BB * 512), blk256, 0, stream, QKV2);
  // 5) channel attention
  hipLaunchKernelGGL(attn_gram, dim3(BB * HEADS), blk256, 0, stream, QKV2, temp, araw);
  hipLaunchKernelGGL(attn_topk_softmax, dim3(BB * HEADS), dim3(1024), 0, stream,
                     araw, dynk, afin);
  hipLaunchKernelGGL(attn_apply, dim3(BB * 256), blk256, 0, stream,
                     afin, QKV2, a1, a2, a3, a4, OATT);
  // 6) interactions + combine + proj
  hipLaunchKernelGGL(chan_mean, dim3(BB * 256), blk256, 0, stream, OATT, cmean);
  hipLaunchKernelGGL(ci_kernel, dim3(BB), blk256, 0, stream, cmean,
                     ci_c1_w, ci_c1_b, ci_bn_g, ci_bn_b, ci_bn_m, ci_bn_v,
                     ci_c2_w, ci_c2_b, cmv);
  hipLaunchKernelGGL(si_kernel, dim3(BB * 16), blk256, 0, stream, YD,
                     si_c1_w, si_c1_b, si_bn_g, si_bn_b, si_bn_m, si_bn_v,
                     si_c2_w, si_c2_b, smap);
  hipLaunchKernelGGL(combine_kernel, dim3(32768), blk256, 0, stream,
                     OATT, YD, smap, cmv, PROJ);
  hipLaunchKernelGGL(gemm_wmma_f16, dim3(64, 8, 8), blk256, 0, stream,
                     projh, PROJ, L, nullptr, 512, 512, st512, st512, 0); // x_out -> L
  // 7) fft2 -> scale -> ifft2 (real)
  const int lines = BB * 512 * 64;
  hipLaunchKernelGGL(dft64, dim3(lines), dim3(64), 0, stream,
                     L, (const float*)nullptr, F1re, F1im, 1, -1.f, 1.f);
  hipLaunchKernelGGL(dft64, dim3(lines), dim3(64), 0, stream,
                     F1re, F1im, F2re, F2im, 0, -1.f, 1.f);
  hipLaunchKernelGGL(fft_scale, dim3(65536), blk256, 0, stream, F2re, F2im, fft_w, fft_b);
  hipLaunchKernelGGL(dft64, dim3(lines), dim3(64), 0, stream,
                     F2re, F2im, F1re, F1im, 0, 1.f, 1.f);
  hipLaunchKernelGGL(dft64, dim3(lines), dim3(64), 0, stream,
                     F1re, F1im, out, (float*)nullptr, 1, 1.f, 1.f / 4096.f);
}
